// DGCNN_Encoder_36112085025205
// MI455X (gfx1250) — matrix operand order repaired and checked by reading, not compiled, not run on hardware
//
#include <hip/hip_runtime.h>
#include <cstdint>
#include <cstddef>

// ---------------------------------------------------------------------------
// DGCNN encoder for MI455X (gfx1250, wave32, WMMA f32_16x16x32_f16)
//   B=8, N=2048, k=20; layers C->O: (3->64)(64->64)(64->128)(128->256), conv5 512->1024
// ---------------------------------------------------------------------------

typedef __attribute__((ext_vector_type(16))) _Float16 v16h;
typedef __attribute__((ext_vector_type(8)))  float    v8f;

#define NB     8
#define NPTS   2048
#define NROWS  (NB * NPTS)      // 16384
#define KNN_K  20
#define NSLICE 64
#define SMAX   1024             // channel stride in the stats slab
#define BN_EPS 1e-5f
#define SLOPE  0.2f

// ---------------- WMMA helpers (A/B: contiguous K = 16*(lane>>4)+s in both; C/D layout per ISA 7.12.2) ----------------

__device__ __forceinline__ v8f wmma16(v16h a, v16h b, v8f c) {
  // (neg_a, A, neg_b, B, c_mod, C, reuse_a, reuse_b)
  return __builtin_amdgcn_wmma_f32_16x16x32_f16(false, a, false, b, (short)0, c,
                                                false, false);
}

// A-matrix 16x32 f16 frag from row-major (stride in halves) storage.
// lane L: M = L&15 ; lanes 0-15 hold K = {0..15}, lanes 16-31 K = {16..31}
__device__ __forceinline__ v16h load_a_frag(const _Float16* base, int stride, int kc) {
  int lane = threadIdx.x & 31;
  const _Float16* row = base + (size_t)(lane & 15) * stride;
  int k0 = kc * 32 + (lane >> 4) * 16;
  union { v16h v; unsigned u[8]; } r;
#pragma unroll
  for (int i = 0; i < 4; ++i) r.u[i]     = *(const unsigned*)(row + k0 + 2 * i);
#pragma unroll
  for (int i = 0; i < 4; ++i) r.u[4 + i] = *(const unsigned*)(row + k0 + 8 + 2 * i);
  return r.v;
}

// B-matrix 32x16 f16 frag; B[k][n] = W[n][k] with W row-major (O x stride halves):
// lane L: N = L&15, K-run = 16 contiguous halves starting at kc*32 + (L>>4)*16.
__device__ __forceinline__ v16h load_b_frag(const _Float16* W, int stride, int obase, int kc) {
  int lane = threadIdx.x & 31;
  const _Float16* p = W + (size_t)(obase + (lane & 15)) * stride + kc * 32 + (lane >> 4) * 16;
  union { v16h v; uint4 q[2]; } r;
  r.q[0] = ((const uint4*)p)[0];
  r.q[1] = ((const uint4*)p)[1];
  return r.v;
}

// monotone float<->u32 encoding for atomicMax on floats
__device__ __forceinline__ unsigned enc_f32(float f) {
  unsigned u = __float_as_uint(f);
  return (u & 0x80000000u) ? ~u : (u | 0x80000000u);
}
__device__ __forceinline__ float dec_f32(unsigned e) {
  return (e & 0x80000000u) ? __uint_as_float(e ^ 0x80000000u) : __uint_as_float(~e);
}

// ---------------- weight convert: fp32 (O x K) -> f16 (O x KP), zero-padded ---

__global__ void convw_kernel(const float* __restrict__ W, _Float16* __restrict__ Wh,
                             int K, int KP, int total) {
  int i = blockIdx.x * blockDim.x + threadIdx.x;
  if (i >= total) return;
  int o = i / KP, k = i - o * KP;
  Wh[i] = (_Float16)(k < K ? W[(size_t)o * K + k] : 0.0f);
}

// ---------------- kNN: one wave per point, top-20 by squared L2 --------------

template <int C>
__global__ __launch_bounds__(128) void knn_kernel(const float* __restrict__ x,
                                                  int* __restrict__ idx) {
  __shared__ unsigned long long keys[4][32 * KNN_K];
  int wid = threadIdx.x >> 5, lane = threadIdx.x & 31;
  int gw = blockIdx.x * 4 + wid;               // point id in [0, B*N)
  int b = gw / NPTS, n = gw - b * NPTS;
  const float* xb = x + (size_t)b * NPTS * C;
  const float* xi = xb + (size_t)n * C;

  unsigned long long top[KNN_K];
#pragma unroll
  for (int t = 0; t < KNN_K; ++t) top[t] = ~0ULL;
  int worst = 0;

  for (int j = lane; j < NPTS; j += 32) {
    const float* xj = xb + (size_t)j * C;
    float d = 0.0f;
#pragma unroll 8
    for (int c = 0; c < C; ++c) { float t = xi[c] - xj[c]; d += t * t; }
    unsigned long long key = ((unsigned long long)__float_as_uint(d) << 32) | (unsigned)j;
    if (key < top[worst]) {
      top[worst] = key;
      unsigned long long mx = 0; int wi = 0;
#pragma unroll
      for (int t = 0; t < KNN_K; ++t) if (top[t] > mx) { mx = top[t]; wi = t; }
      worst = wi;
    }
  }
#pragma unroll
  for (int t = 0; t < KNN_K; ++t) keys[wid][lane * KNN_K + t] = top[t];
  __syncthreads();
  if (lane == 0) {                              // merge 640 candidates -> best 20
    int* out = idx + (size_t)gw * KNN_K;
    for (int t = 0; t < KNN_K; ++t) {
      unsigned long long best = ~0ULL; int bi = 0;
      for (int e = 0; e < 32 * KNN_K; ++e) {
        unsigned long long k = keys[wid][e];
        if (k < best) { best = k; bi = e; }
      }
      keys[wid][bi] = ~0ULL;
      out[t] = (int)(best & 0xffffffffu);
    }
  }
}

// ---------------- fused EdgeConv GEMM: gather -> WMMA -> k-max + BN stats ----
// one block per (b,n); 4 waves; feat rows 0..19 = [x_j - x_n , x_n], 20..31 = 0

template <int C, int O, int KP>
__global__ __launch_bounds__(128) void edge_gemm_kernel(
    const float* __restrict__ xin, const int* __restrict__ idx,
    const _Float16* __restrict__ Wh, float* __restrict__ hmax_out,
    float* __restrict__ sums) {
  __shared__ _Float16 feat[32][KP];
  __shared__ float    hsm[32][O];

  int bn = blockIdx.x;
  int b = bn / NPTS, n = bn - b * NPTS;
  const float* xb = xin + (size_t)b * NPTS * C;
  const float* xc = xb + (size_t)n * C;
  const int*   id = idx + (size_t)bn * KNN_K;
  int tid = threadIdx.x;

  for (int e = tid; e < 32 * KP; e += 128) {
    int t = e / KP, k = e - t * KP;
    float v = 0.0f;
    if (t < KNN_K) {
      if (k < C)            { int j = id[t]; v = xb[(size_t)j * C + k] - xc[k]; }
      else if (k < 2 * C)   { v = xc[k - C]; }
    }
    feat[t][k] = (_Float16)v;
  }
  __syncthreads();

  const int ONT = O / 16;
  int w = tid >> 5, lane = tid & 31;
  for (int tile = w; tile < 2 * ONT; tile += 4) {
    int mt = tile / ONT, nt = tile - mt * ONT;
    union { v8f v; float f[8]; } acc; acc.v = (v8f){};
#pragma unroll
    for (int kc = 0; kc < KP / 32; ++kc) {
      v16h a  = load_a_frag(&feat[mt * 16][0], KP, kc);
      v16h bf = load_b_frag(Wh, KP, nt * 16, kc);
      acc.v = wmma16(a, bf, acc.v);
    }
    int mofs = (lane < 16) ? 0 : 8;             // C/D layout: M = v + 8*(lane>=16)
    int col = nt * 16 + (lane & 15);
#pragma unroll
    for (int v = 0; v < 8; ++v) hsm[mt * 16 + mofs + v][col] = acc.f[v];
  }
  __syncthreads();

  int slice = bn & (NSLICE - 1);
  for (int o = tid; o < O; o += 128) {
    float mx = -3.4e38f, s = 0.0f, s2 = 0.0f;
#pragma unroll 4
    for (int t = 0; t < KNN_K; ++t) {
      float v = hsm[t][o];
      mx = fmaxf(mx, v); s += v; s2 += v * v;
    }
    hmax_out[(size_t)bn * O + o] = mx;
    atomicAdd(&sums[slice * 2 * SMAX + o],        s);
    atomicAdd(&sums[slice * 2 * SMAX + SMAX + o], s2);
  }
}

// ---------------- BN stat finalize + apply (max commutes: gamma > 0) ---------

__global__ void finalize_stats_kernel(const float* __restrict__ sums,
                                      float* __restrict__ stats, int O, float inv_cnt) {
  int o = blockIdx.x * blockDim.x + threadIdx.x;
  if (o >= O) return;
  float s = 0.0f, s2 = 0.0f;
  for (int sl = 0; sl < NSLICE; ++sl) {
    s  += sums[sl * 2 * SMAX + o];
    s2 += sums[sl * 2 * SMAX + SMAX + o];
  }
  float mean = s * inv_cnt;
  float var  = fmaxf(s2 * inv_cnt - mean * mean, 0.0f);
  stats[o]        = mean;
  stats[SMAX + o] = rsqrtf(var + BN_EPS);
}

__global__ void apply_bn_kernel(float* __restrict__ x, const float* __restrict__ stats,
                                const float* __restrict__ gamma, const float* __restrict__ beta,
                                int O, int total) {
  int i = blockIdx.x * blockDim.x + threadIdx.x;
  if (i >= total) return;
  int o = i % O;
  float v = (x[i] - stats[o]) * stats[SMAX + o] * gamma[o] + beta[o];
  x[i] = (v >= 0.0f) ? v : SLOPE * v;
}

// ---------------- conv5: concat -> f16, tiled WMMA GEMM + fused max/stats ----

__global__ void concat_kernel(const float* __restrict__ x1, const float* __restrict__ x2,
                              const float* __restrict__ x3, const float* __restrict__ x4,
                              _Float16* __restrict__ xch) {
  int i = blockIdx.x * blockDim.x + threadIdx.x;
  if (i >= NROWS * 512) return;
  int r = i >> 9, c = i & 511;
  float v;
  if (c < 64)       v = x1[(size_t)r * 64  + c];
  else if (c < 128) v = x2[(size_t)r * 64  + (c - 64)];
  else if (c < 256) v = x3[(size_t)r * 128 + (c - 128)];
  else              v = x4[(size_t)r * 256 + (c - 256)];
  xch[i] = (_Float16)v;
}

__global__ __launch_bounds__(128) void conv5_gemm_kernel(
    const _Float16* __restrict__ A,      // (16384 x 512) f16
    const _Float16* __restrict__ Wh,     // (1024 x 512) f16 (= B^T rows)
    float* __restrict__ sums, unsigned* __restrict__ umax) {
  __shared__ _Float16 atile[16][512];
  int mt = blockIdx.x;                   // 0..1023 (16 rows each)
  int ntb = blockIdx.y;                  // 0..15  (64 cols each)
  int tid = threadIdx.x, w = tid >> 5, lane = tid & 31;

  const _Float16* arow = A + (size_t)mt * 16 * 512;
  for (int e = tid; e < 1024; e += 128)  // 16*512 halves = 1024 uint4
    ((uint4*)&atile[0][0])[e] = ((const uint4*)arow)[e];
  __syncthreads();

  int nt = ntb * 4 + w;
  union { v8f v; float f[8]; } acc; acc.v = (v8f){};
#pragma unroll 4
  for (int kc = 0; kc < 16; ++kc) {
    v16h a  = load_a_frag(&atile[0][0], 512, kc);
    v16h bf = load_b_frag(Wh, 512, nt * 16, kc);
    acc.v = wmma16(a, bf, acc.v);
  }
  int o = nt * 16 + (lane & 15);
  float s = 0.0f, s2 = 0.0f, mx = -3.4e38f;
#pragma unroll
  for (int v = 0; v < 8; ++v) { float x = acc.f[v]; s += x; s2 += x * x; mx = fmaxf(mx, x); }
  s  += __shfl_xor(s, 16, 32);           // lanes L and L^16 hold same column
  s2 += __shfl_xor(s2, 16, 32);
  mx  = fmaxf(mx, __shfl_xor(mx, 16, 32));
  if (lane < 16) {
    int slice = (mt + ntb) & (NSLICE - 1);
    atomicAdd(&sums[slice * 2 * SMAX + o],        s);
    atomicAdd(&sums[slice * 2 * SMAX + SMAX + o], s2);
    int b = mt >> 7;                     // 128 row-tiles per batch
    atomicMax(&umax[b * SMAX + o], enc_f32(mx));
  }
}

__global__ void conv5_out_kernel(const float* __restrict__ sums,
                                 const unsigned* __restrict__ umax,
                                 const float* __restrict__ g, const float* __restrict__ bt,
                                 float* __restrict__ out) {
  int o = blockIdx.x * blockDim.x + threadIdx.x;
  if (o >= SMAX) return;
  float s = 0.0f, s2 = 0.0f;
  for (int sl = 0; sl < NSLICE; ++sl) {
    s  += sums[sl * 2 * SMAX + o];
    s2 += sums[sl * 2 * SMAX + SMAX + o];
  }
  const float inv_cnt = 1.0f / (float)NROWS;
  float mean = s * inv_cnt;
  float var  = fmaxf(s2 * inv_cnt - mean * mean, 0.0f);
  float istd = rsqrtf(var + BN_EPS);
  for (int b = 0; b < NB; ++b) {
    float v = (dec_f32(umax[b * SMAX + o]) - mean) * istd * g[o] + bt[o];
    out[b * SMAX + o] = (v >= 0.0f) ? v : SLOPE * v;
  }
}

// ---------------------------------------------------------------------------

extern "C" void kernel_launch(void* const* d_in, const int* in_sizes, int n_in,
                              void* d_out, int out_size, void* d_ws, size_t ws_size,
                              hipStream_t stream) {
  (void)in_sizes; (void)n_in; (void)out_size; (void)ws_size;
  const float* x  = (const float*)d_in[0];
  const float* W1 = (const float*)d_in[1];
  const float* g1 = (const float*)d_in[2];  const float* b1 = (const float*)d_in[3];
  const float* W2 = (const float*)d_in[4];
  const float* g2 = (const float*)d_in[5];  const float* b2 = (const float*)d_in[6];
  const float* W3 = (const float*)d_in[7];
  const float* g3 = (const float*)d_in[8];  const float* b3 = (const float*)d_in[9];
  const float* W4 = (const float*)d_in[10];
  const float* g4 = (const float*)d_in[11]; const float* b4 = (const float*)d_in[12];
  const float* W5 = (const float*)d_in[13];
  const float* g5 = (const float*)d_in[14]; const float* b5 = (const float*)d_in[15];
  float* out = (float*)d_out;

  // workspace carving (256B aligned)
  char*  base = (char*)d_ws;
  size_t off = 0;
  auto carve = [&](size_t bytes) -> char* {
    char* p = base + off;
    off = (off + bytes + 255) & ~(size_t)255;
    return p;
  };
  int*       idx  = (int*)      carve((size_t)NROWS * KNN_K * 4);
  float*     x1   = (float*)    carve((size_t)NROWS * 64 * 4);
  float*     x2   = (float*)    carve((size_t)NROWS * 64 * 4);
  float*     x3   = (float*)    carve((size_t)NROWS * 128 * 4);
  float*     x4   = (float*)    carve((size_t)NROWS * 256 * 4);
  _Float16*  Wh1  = (_Float16*) carve((size_t)64 * 32 * 2);
  _Float16*  Wh2  = (_Float16*) carve((size_t)64 * 128 * 2);
  _Float16*  Wh3  = (_Float16*) carve((size_t)128 * 128 * 2);
  _Float16*  Wh4  = (_Float16*) carve((size_t)256 * 256 * 2);
  _Float16*  Wh5  = (_Float16*) carve((size_t)1024 * 512 * 2);
  _Float16*  xch  = (_Float16*) carve((size_t)NROWS * 512 * 2);
  float*     sums = (float*)    carve((size_t)NSLICE * 2 * SMAX * 4);
  float*     stats= (float*)    carve((size_t)2 * SMAX * 4);
  unsigned*  umax = (unsigned*) carve((size_t)NB * SMAX * 4);

  const size_t sums_bytes = (size_t)NSLICE * 2 * SMAX * 4;
  const float inv_cnt_e = 1.0f / ((float)NROWS * KNN_K);

  // weights -> f16 (zero-padded K)
  auto cgrid = [](int total) { return dim3((unsigned)((total + 255) / 256)); };
  convw_kernel<<<cgrid(64 * 32),    256, 0, stream>>>(W1, Wh1, 6,   32,  64 * 32);
  convw_kernel<<<cgrid(64 * 128),   256, 0, stream>>>(W2, Wh2, 128, 128, 64 * 128);
  convw_kernel<<<cgrid(128 * 128),  256, 0, stream>>>(W3, Wh3, 128, 128, 128 * 128);
  convw_kernel<<<cgrid(256 * 256),  256, 0, stream>>>(W4, Wh4, 256, 256, 256 * 256);
  convw_kernel<<<cgrid(1024 * 512), 256, 0, stream>>>(W5, Wh5, 512, 512, 1024 * 512);

  dim3 knn_grid(NROWS / 4), blk128(128);

  // ---- layer 1: 3 -> 64
  knn_kernel<3><<<knn_grid, blk128, 0, stream>>>(x, idx);
  hipMemsetAsync(sums, 0, sums_bytes, stream);
  edge_gemm_kernel<3, 64, 32><<<dim3(NROWS), blk128, 0, stream>>>(x, idx, Wh1, x1, sums);
  finalize_stats_kernel<<<dim3(1), dim3(64), 0, stream>>>(sums, stats, 64, inv_cnt_e);
  apply_bn_kernel<<<cgrid(NROWS * 64), 256, 0, stream>>>(x1, stats, g1, b1, 64, NROWS * 64);

  // ---- layer 2: 64 -> 64
  knn_kernel<64><<<knn_grid, blk128, 0, stream>>>(x1, idx);
  hipMemsetAsync(sums, 0, sums_bytes, stream);
  edge_gemm_kernel<64, 64, 128><<<dim3(NROWS), blk128, 0, stream>>>(x1, idx, Wh2, x2, sums);
  finalize_stats_kernel<<<dim3(1), dim3(64), 0, stream>>>(sums, stats, 64, inv_cnt_e);
  apply_bn_kernel<<<cgrid(NROWS * 64), 256, 0, stream>>>(x2, stats, g2, b2, 64, NROWS * 64);

  // ---- layer 3: 64 -> 128
  knn_kernel<64><<<knn_grid, blk128, 0, stream>>>(x2, idx);
  hipMemsetAsync(sums, 0, sums_bytes, stream);
  edge_gemm_kernel<64, 128, 128><<<dim3(NROWS), blk128, 0, stream>>>(x2, idx, Wh3, x3, sums);
  finalize_stats_kernel<<<dim3(1), dim3(128), 0, stream>>>(sums, stats, 128, inv_cnt_e);
  apply_bn_kernel<<<cgrid(NROWS * 128), 256, 0, stream>>>(x3, stats, g3, b3, 128, NROWS * 128);

  // ---- layer 4: 128 -> 256
  knn_kernel<128><<<knn_grid, blk128, 0, stream>>>(x3, idx);
  hipMemsetAsync(sums, 0, sums_bytes, stream);
  edge_gemm_kernel<128, 256, 256><<<dim3(NROWS), blk128, 0, stream>>>(x3, idx, Wh4, x4, sums);
  finalize_stats_kernel<<<dim3(2), dim3(128), 0, stream>>>(sums, stats, 256, inv_cnt_e);
  apply_bn_kernel<<<cgrid(NROWS * 256), 256, 0, stream>>>(x4, stats, g4, b4, 256, NROWS * 256);

  // ---- conv5: concat(512) -> 1024, BN over (B,N), global max over N
  concat_kernel<<<cgrid(NROWS * 512), 256, 0, stream>>>(x1, x2, x3, x4, xch);
  hipMemsetAsync(sums, 0, sums_bytes, stream);
  hipMemsetAsync(umax, 0, (size_t)NB * SMAX * 4, stream);
  conv5_gemm_kernel<<<dim3(NROWS / 16, 16), blk128, 0, stream>>>(xch, Wh5, sums, umax);
  conv5_out_kernel<<<dim3(8), blk128, 0, stream>>>(sums, umax, g5, b5, out);
}
